// RGB_Conv2d_58351425683924
// MI455X (gfx1250) — matrix operand order repaired and checked by reading, not compiled, not run on hardware
//
#include <hip/hip_runtime.h>
#include <hip/hip_bf16.h>

typedef __attribute__((ext_vector_type(16))) _Float16 v16h;
typedef __attribute__((ext_vector_type(8)))  _Float16 v8h;
typedef __attribute__((ext_vector_type(8)))  float    v8f;
typedef __attribute__((ext_vector_type(4)))  float    v4f;

#define IH 224
#define IW 224
#define OH 220
#define OW 220
#define NCOL 30
#define NBAT 8

// LDS region geometry
#define DROWS 20
#define DCOLS 68
#define DSTR  72   // f32 distance map stride (words)
#define VROWS 16
#define VCOLS 68
#define VSTR  88   // f16 colsum stride (halves)
#define SOSTR 68   // f32 output staging stride (words); offsets hi/lo half-wave banks

__global__ __launch_bounds__(128) void rgb_conv2d_kernel(
    const float* __restrict__ in,   // [8,3,224,224]
    const float* __restrict__ wt,   // [30,3,5,5]
    float* __restrict__ out)        // [8,30,220,220]
{
  // sDO: stage 0 distance map (20x72 f32), reused after stage 1 as the
  //      16x68 f32 output staging tile (both never live at the same time).
  __shared__ __align__(16) float    sDO[DROWS * DSTR];
  __shared__ __align__(16) _Float16 sV[VROWS * VSTR];

  const int tid = threadIdx.x;
  const int bx  = blockIdx.x;        // 0..55 = ty*4 + tg
  const int c   = blockIdx.y;        // color 0..29
  const int b   = blockIdx.z;        // batch 0..7
  const int ty  = bx >> 2;
  const int tg  = bx & 3;
  const int y0  = ty * 16;           // first output row of this block
  const int x0g = tg * 64;           // first output col of this block

  // Prototype color, pre-scaled by 255 (weight is color/255 tiled over 5x5).
  const float R2 = wt[(c * 3 + 0) * 25] * 255.0f;
  const float G2 = wt[(c * 3 + 1) * 25] * 255.0f;
  const float B2 = wt[(c * 3 + 2) * 25] * 255.0f;

  const float* pR = in + (size_t)(b * 3 + 0) * (IH * IW);
  const float* pG = in + (size_t)(b * 3 + 1) * (IH * IW);
  const float* pB = in + (size_t)(b * 3 + 2) * (IH * IW);

  // ---- Stage 0: per-pixel color distance into LDS (rows y0..y0+19, cols x0g..x0g+67) ----
  for (int idx = tid; idx < DROWS * DCOLS; idx += 128) {
    int r = idx / DCOLS;
    int q = idx - r * DCOLS;
    int gy = y0 + r;  if (gy > IH - 1) gy = IH - 1;   // clamp: OOB outputs masked at store
    int gx = x0g + q; if (gx > IW - 1) gx = IW - 1;
    int off = gy * IW + gx;
    float R1 = pR[off] * 255.0f;
    float G1 = pG[off] * 255.0f;
    float B1 = pB[off] * 255.0f;
    float rmean = (R1 + R2) * 0.5f;
    float dR = R1 - R2, dG = G1 - G2, dB = B1 - B2;
    float t = (2.0f + rmean * (1.0f / 256.0f)) * (dR * dR)
            + 4.0f * (dG * dG)
            + (2.0f + (255.0f - rmean) * (1.0f / 256.0f)) * (dB * dB)
            + 1e-8f;
    sDO[r * DSTR + q] = sqrtf(t);
  }
  __syncthreads();

  // ---- Stage 1: vertical 5-sum -> f16 LDS; cols [VCOLS,VSTR) zero (finite padding for K>=20) ----
  for (int idx = tid; idx < VROWS * VSTR; idx += 128) {
    int m = idx / VSTR;
    int q = idx - m * VSTR;
    float s = 0.0f;
    if (q < VCOLS) {
      const float* p = &sDO[m * DSTR + q];
      s = p[0] + p[DSTR] + p[2 * DSTR] + p[3 * DSTR] + p[4 * DSTR];
    }
    sV[m * VSTR + q] = (_Float16)s;
  }
  __syncthreads();   // also: all reads of the distance map done -> sDO reusable

  // ---- Stage 2: horizontal 5-sum as WMMA: Out(16x16) = Dv(16x32,f16) x Band(32x16,f16) ----
  const int wv   = tid >> 5;        // wave id = x-tile within the 64-col group
  const int lane = tid & 31;
  const int nrow = lane & 15;       // A: row M; B/D: column N
  const int hi   = lane >> 4;

  // A layout (16-bit 16x32): lane holds K = hi*8 + e (e<8) and 16 + hi*8 + (e-8).
  const int col0 = (wv << 4) + (hi << 3);  // Dv column of K=hi*8 for this tile
  union { v16h v; v8h h[2]; } A;
  A.h[0] = *(const v8h*)&sV[nrow * VSTR + col0];
  A.h[1] = *(const v8h*)&sV[nrow * VSTR + col0 + 16];

  // B layout (32x16): lane holds K = hi*8 + e (e<8) and 16 + hi*8 + (e-8), N=lane&15.
  // Band of ones: S[k][n] = 1 iff n <= k <= n+4  (k<=19 automatic since n<=15).
  v16h Bm;
  const int kb = hi << 3;
#pragma unroll
  for (int e = 0; e < 16; ++e) {
    int K = kb + e + ((e >= 8) ? 8 : 0);
    Bm[e] = (K >= nrow && K <= nrow + 4) ? (_Float16)1.0f : (_Float16)0.0f;
  }

  v8f acc = {};
  acc = __builtin_amdgcn_wmma_f32_16x16x32_f16(
      /*neg_a=*/false, A.v, /*neg_b=*/false, Bm,
      /*c_mod=*/(short)0, acc, /*reuse_a=*/false, /*reuse_b=*/false);

  // ---- Stage 3: stage result in LDS, then coalesced float4 non-temporal stores ----
  // D layout: VGPR v -> tile row hi*8 + v, tile column 16*wv + nrow.
  float* sOut = sDO;
#pragma unroll
  for (int v = 0; v < 8; ++v)
    sOut[((hi << 3) + v) * SOSTR + (wv << 4) + nrow] = acc[v] * (1.0f / 19200.0f);
  __syncthreads();

  // 16 rows x 64 cols = 256 float4s; rows of out are 55 float4s (16B-aligned),
  // last col group (tg==3) holds exactly 7 valid float4s (cols 192..219).
  const int n4 = (tg < 3) ? 16 : 7;
  float* obase = out + (size_t)(b * NCOL + c) * OH * OW + x0g;
  for (int idx = tid; idx < 16 * 16; idx += 128) {
    int row = idx >> 4;
    int q4  = idx & 15;
    int gy  = y0 + row;
    if (q4 < n4 && gy < OH) {
      v4f val = *(const v4f*)&sOut[row * SOSTR + (q4 << 2)];
      __builtin_nontemporal_store(val, (v4f*)(obase + (size_t)gy * OW + (q4 << 2)));
    }
  }
}

extern "C" void kernel_launch(void* const* d_in, const int* in_sizes, int n_in,
                              void* d_out, int out_size, void* d_ws, size_t ws_size,
                              hipStream_t stream) {
  (void)in_sizes; (void)n_in; (void)d_ws; (void)ws_size; (void)out_size;
  const float* in = (const float*)d_in[0];   // [8,3,224,224] f32
  const float* wt = (const float*)d_in[1];   // [30,3,5,5]    f32
  float* out = (float*)d_out;                // [8,30,220,220] f32

  dim3 grid(56, NCOL, NBAT);   // 14 row-tiles * 4 col-groups, 30 colors, 8 batches
  dim3 block(128);             // 4 wave32 -> one 16x16 WMMA tile per wave
  hipLaunchKernelGGL(rgb_conv2d_kernel, grid, block, 0, stream, in, wt, out);
}
